// LocationSlayerArch_6408091205838
// MI455X (gfx1250) — matrix operand order repaired and checked by reading, not compiled, not run on hardware
//
#include <hip/hip_runtime.h>
#include <hip/hip_bf16.h>

// ---------------------------------------------------------------------------
// SLAYER SRM-alpha SNN for MI455X (gfx1250, wave32, WMMA).
//
// Restructured by linearity:  psp∘dense == dense∘psp, so
//   layer1: U1 = W1 @ psp(X)        (psp on the cheap 156-ch tensor)
//   layer2: U2 = psp(W2 @ S1)       (dense first: 1024 -> 20 channels)
// Big GEMMs use v_wmma_f32_16x16x32_bf16, 5 output tiles per wave (16x80),
// double-buffered LDS B staging. Iteration order: load A + all 5 B frags,
// 5 back-to-back WMMAs, then stage next tile (chunked: 8 independent global
// loads -> one ds_store_b128), one barrier. Spike refractory scan:
// register-resident 16-deep shift per neuron.
// ---------------------------------------------------------------------------

typedef __attribute__((ext_vector_type(16))) __bf16 v16bf;
typedef __attribute__((ext_vector_type(8)))  __bf16 v8bf;
typedef __attribute__((ext_vector_type(8)))  float  v8f;

#define TSAMP 156
#define TPAD  160
#define HID   1024
#define NOUT  20
#define MPAD2 32
#define NB    128
#define THETA 10.0f

#define JT    5                   // j-tiles (of 16 cols) per wave
#define JCOLS (JT * 16)           // 80 columns per block
#define TILE_ELEMS (JCOLS * 32)   // B tile: 32 k-rows x 80 cols
#define TILE_CHUNKS (TILE_ELEMS / 8)  // 320 chunks of 8 k-values

__device__ __constant__ int c_top_down[39] = {
    11,25,35,4,18,30,7,2,20,37,29,12,9,33,23,16,1,6,15,21,
    27,34,39,24,17,10,31,38,28,14,3,22,32,8,19,36,5,13,26};

__device__ __forceinline__ int perm_idx(int s) {
    int base = (s < 78) ? 0 : 78;
    int s2 = s - base;
    return 2 * c_top_down[s2 >> 1] - 2 + (s2 & 1) + base;
}

__device__ __forceinline__ float srm_kern(int d) {
    float td = (float)d;
    return td * 0.1f * __expf(1.0f - td * 0.1f);
}

// stage one chunk: column c, k-values k8..k8+7 -> BT[c*32+k8..+7]
__device__ __forceinline__ void stage_chunk(const __bf16* __restrict__ Bsrc,
                                            __bf16* __restrict__ dst,
                                            int Jpad, int ch) {
    const int c  = ch >> 2;          // 0..79
    const int k8 = (ch & 3) * 8;     // 0,8,16,24
    const __bf16* g = Bsrc + (size_t)k8 * Jpad + c;
    v8bf tmp;
#pragma unroll
    for (int i = 0; i < 8; ++i) tmp[i] = g[(size_t)i * Jpad];  // 8 loads in flight
    *(v8bf*)(&dst[c * 32 + k8]) = tmp;                         // one ds_store_b128
}

// ---- weight fp32 -> bf16 with zero padding -------------------------------
__global__ void convert_w_kernel(const float* __restrict__ src,
                                 __bf16* __restrict__ dst,
                                 int M, int K, int Kpad) {
    int m = blockIdx.x;  // 0..Mpad-1
    for (int k = threadIdx.x; k < Kpad; k += blockDim.x) {
        float v = (m < M && k < K) ? src[(size_t)m * K + k] : 0.0f;
        dst[(size_t)m * Kpad + k] = (__bf16)v;
    }
}

// ---- P = psp(X) along time, main path: P[n, c, t] ------------------------
__global__ void pack_psp_main_kernel(const float* __restrict__ x,
                                     __bf16* __restrict__ P) {
    const int c = blockIdx.x;   // 0..159 (rows >=156 are K-pad -> zero)
    const int n = blockIdx.y;
    const int t = threadIdx.x;  // 0..159
    __shared__ float xrow[TPAD];
    __shared__ float kern[TPAD];
    if (t < TSAMP) {
        kern[t] = srm_kern(t);
        xrow[t] = (c < TSAMP) ? x[((size_t)n * TSAMP + c) * TSAMP + t] : 0.0f;
    }
    __syncthreads();
    float acc = 0.0f;
    if (c < TSAMP && t < TSAMP) {
        for (int s = 0; s <= t; ++s) acc += xrow[s] * kern[t - s];
    }
    P[((size_t)n * TPAD + c) * TPAD + t] = (__bf16)acc;
}

// ---- location path: P[n, a, c'] = sum_{s<=c'} x[n,PERM[s],a]*kern[c'-s] --
__global__ void pack_psp_loc_kernel(const float* __restrict__ x,
                                    __bf16* __restrict__ P) {
    const int a = blockIdx.x;   // original time index (GEMM reduction row)
    const int n = blockIdx.y;
    const int t = threadIdx.x;  // c' (permuted taxel = psp axis)
    __shared__ float xrow[TPAD];
    __shared__ float kern[TPAD];
    if (t < TSAMP) {
        kern[t] = srm_kern(t);
        xrow[t] = (a < TSAMP) ? x[((size_t)n * TSAMP + perm_idx(t)) * TSAMP + a] : 0.0f;
    }
    __syncthreads();
    float acc = 0.0f;
    if (a < TSAMP && t < TSAMP) {
        for (int s = 0; s <= t; ++s) acc += xrow[s] * kern[t - s];
    }
    P[((size_t)n * TPAD + a) * TPAD + t] = (__bf16)acc;
}

// ---- batched WMMA GEMM: C[n] = A @ B[n]  (bf16 x bf16 -> f32) ------------
// A: [Mtot x K] row-major (K = lda, multiple of 32)
// B: [batch][K x Jpad], C: [batch][Mtot x Jpad], Jpad multiple of 80.
// Each wave computes a 16 x 80 strip; BLOCK/32 waves stack along M.
template <int BLOCK>
__global__ void wmma_gemm_bf16_kernel(const __bf16* __restrict__ A,
                                      const __bf16* __restrict__ B,
                                      float* __restrict__ C,
                                      int K, int Jpad,
                                      size_t strideB, size_t strideC) {
    constexpr int WPB = BLOCK / 32;

    const int wave = threadIdx.x >> 5;
    const int lane = threadIdx.x & 31;
    const int hi   = lane >> 4;     // half-wave select
    const int lo   = lane & 15;
    const int mbase = (blockIdx.x * WPB + wave) * 16;
    const int jbase = blockIdx.y * JCOLS;
    const int n     = blockIdx.z;

    const __bf16* Bn = B + (size_t)n * strideB;
    float*        Cn = C + (size_t)n * strideC;

    // B tiles staged transposed: BT[buf][col][k]; each lane's fragment is two
    // 16-byte ds reads (k = hi*8.., 16+hi*8..). Double-buffered (ping-pong).
    __shared__ __align__(32) __bf16 BT[2][TILE_ELEMS];

    v8f acc[JT];
#pragma unroll
    for (int jt = 0; jt < JT; ++jt)
#pragma unroll
        for (int i = 0; i < 8; ++i) acc[jt][i] = 0.0f;

    // prime buffer 0 with k-tile 0
    for (int ch = threadIdx.x; ch < TILE_CHUNKS; ch += BLOCK)
        stage_chunk(Bn + jbase, BT[0], Jpad, ch);
    __syncthreads();

    int cur = 0;
    for (int kk = 0; kk < K; kk += 32) {
        // ---- 1) A fragment (documented 16-bit 16x32 layout):
        // lane(lo,hi) row m=mbase+lo, halves i: k = (i&7) + hi*8 + (i>>3)*16
        const __bf16* Ap = A + (size_t)(mbase + lo) * K + kk + hi * 8;
        v8bf alo = *(const v8bf*)Ap;          // kk+hi*8    .. +7  (b128)
        v8bf ahi = *(const v8bf*)(Ap + 16);   // kk+16+hi*8 .. +7  (b128)
        v16bf a;
#pragma unroll
        for (int i = 0; i < 8; ++i) { a[i] = alo[i]; a[i + 8] = ahi[i]; }

        // ---- 2) all 5 B fragments (ds loads pipeline under partial waits)
        v16bf b[JT];
#pragma unroll
        for (int jt = 0; jt < JT; ++jt) {
            const __bf16* bp = &BT[cur][(jt * 16 + lo) * 32 + hi * 8];
            v8bf blo = *(const v8bf*)bp;
            v8bf bhi = *(const v8bf*)(bp + 16);
#pragma unroll
            for (int i = 0; i < 8; ++i) { b[jt][i] = blo[i]; b[jt][i + 8] = bhi[i]; }
        }

        // ---- 3) five back-to-back WMMAs (independent accumulators)
#pragma unroll
        for (int jt = 0; jt < JT; ++jt)
            acc[jt] = __builtin_amdgcn_wmma_f32_16x16x32_bf16(
                false, a, false, b[jt], (short)0, acc[jt], false, false);

        // ---- 4) stage next k-tile while the matrix pipe drains
        if (kk + 32 < K) {
            const __bf16* Bnext = Bn + (size_t)(kk + 32) * Jpad + jbase;
            __bf16* dst = BT[cur ^ 1];
            for (int ch = threadIdx.x; ch < TILE_CHUNKS; ch += BLOCK)
                stage_chunk(Bnext, dst, Jpad, ch);
        }
        if (kk + 64 < K) {
            __builtin_prefetch(&Bn[(size_t)(kk + 64) * Jpad + jbase + lo], 0, 1);
        }

        __syncthreads();   // next-buffer writes visible; cur safe to recycle
        cur ^= 1;
    }

    // D layout: VGPR r -> row r (lanes 0-15) / row 8+r (lanes 16-31), col = lo
#pragma unroll
    for (int jt = 0; jt < JT; ++jt) {
#pragma unroll
        for (int r = 0; r < 8; ++r) {
            Cn[(size_t)(mbase + r + hi * 8) * Jpad + jbase + jt * 16 + lo] =
                acc[jt][r];
        }
    }
}

// ---- spike scan with 16-deep refractory pending buffer -------------------
__global__ void spike_scan_kernel(const float* __restrict__ U,
                                  __bf16* __restrict__ S, int total) {
    int r = blockIdx.x * blockDim.x + threadIdx.x;
    if (r >= total) return;
    const float* u = U + (size_t)r * TPAD;
    __bf16*      s = S + (size_t)r * TPAD;

    float refn[16];
#pragma unroll
    for (int d = 0; d < 16; ++d) {
        float td = (float)(d + 1);
        refn[d] = (d < 15) ? (-20.0f * td * __expf(1.0f - td)) : 0.0f;
    }
    float pend[16];
#pragma unroll
    for (int d = 0; d < 16; ++d) pend[d] = 0.0f;

    for (int t = 0; t < TSAMP; ++t) {
        float um = u[t] + pend[0];
        float sp = (um >= THETA) ? 1.0f : 0.0f;
#pragma unroll
        for (int d = 0; d < 15; ++d) pend[d] = pend[d + 1] + sp * refn[d];
        pend[15] = 0.0f;
        s[t] = (__bf16)sp;
    }
    // zero the J-pad columns
    for (int t = TSAMP; t < TPAD; ++t) s[t] = (__bf16)0.0f;
}

// ---- small psp + spike scan + write into concatenated output -------------
__global__ void psp_spike_out_kernel(const float* __restrict__ V,
                                     float* __restrict__ out, int toff) {
    const int o = blockIdx.x;   // 0..19
    const int n = blockIdx.y;   // 0..127
    const int t = threadIdx.x;  // 0..159
    __shared__ float vrow[TPAD];
    __shared__ float kern[TPAD];
    __shared__ float u2[TPAD];
    if (t < TSAMP) {
        kern[t] = srm_kern(t);
        vrow[t] = V[((size_t)n * MPAD2 + o) * TPAD + t];
    }
    __syncthreads();
    if (t < TSAMP) {
        float acc = 0.0f;
        for (int s = 0; s <= t; ++s) acc += vrow[s] * kern[t - s];
        u2[t] = acc;
    }
    __syncthreads();
    if (t == 0) {
        float refn[16];
#pragma unroll
        for (int d = 0; d < 16; ++d) {
            float td = (float)(d + 1);
            refn[d] = (d < 15) ? (-20.0f * td * __expf(1.0f - td)) : 0.0f;
        }
        float pend[16];
#pragma unroll
        for (int d = 0; d < 16; ++d) pend[d] = 0.0f;
        float* op = out + ((size_t)n * NOUT + o) * (2 * TSAMP) + toff;
        for (int tt = 0; tt < TSAMP; ++tt) {
            float um = u2[tt] + pend[0];
            float sp = (um >= THETA) ? 1.0f : 0.0f;
#pragma unroll
            for (int d = 0; d < 15; ++d) pend[d] = pend[d + 1] + sp * refn[d];
            pend[15] = 0.0f;
            op[tt] = sp;
        }
    }
}

// ---------------------------------------------------------------------------
extern "C" void kernel_launch(void* const* d_in, const int* in_sizes, int n_in,
                              void* d_out, int out_size, void* d_ws, size_t ws_size,
                              hipStream_t stream) {
    (void)in_sizes; (void)n_in; (void)out_size; (void)ws_size;
    const float* x      = (const float*)d_in[0];
    const float* w_fc1  = (const float*)d_in[1];
    const float* w_fc2  = (const float*)d_in[2];
    const float* w_loc1 = (const float*)d_in[3];
    const float* w_loc2 = (const float*)d_in[4];
    float* out = (float*)d_out;

    char* ws = (char*)d_ws;
    size_t off = 0;
    auto take = [&](size_t bytes) -> void* {
        void* p = ws + off;
        off = (off + bytes + 255) & ~(size_t)255;
        return p;
    };
    __bf16* W1b = (__bf16*)take((size_t)HID * TPAD * 2);                 // 320 KB
    __bf16* W2b = (__bf16*)take((size_t)MPAD2 * HID * 2);                // 64 KB
    __bf16* P   = (__bf16*)take((size_t)NB * TPAD * TPAD * 2);           // 6.6 MB
    float*  U1  = (float*) take((size_t)NB * HID * TPAD * 4);            // 84 MB
    __bf16* S1  = (__bf16*)take((size_t)NB * HID * TPAD * 2);            // 42 MB
    float*  V   = (float*) take((size_t)NB * MPAD2 * TPAD * 4);          // 2.6 MB

    for (int path = 0; path < 2; ++path) {
        const float* wa = path ? w_loc1 : w_fc1;
        const float* wb = path ? w_loc2 : w_fc2;

        convert_w_kernel<<<HID, 256, 0, stream>>>(wa, W1b, HID, TSAMP, TPAD);
        convert_w_kernel<<<MPAD2, 256, 0, stream>>>(wb, W2b, NOUT, HID, HID);

        if (path == 0)
            pack_psp_main_kernel<<<dim3(TPAD, NB), TPAD, 0, stream>>>(x, P);
        else
            pack_psp_loc_kernel<<<dim3(TPAD, NB), TPAD, 0, stream>>>(x, P);

        // U1[n] = W1b(1024x160) @ P[n](160x160):
        // 8 waves/block along M, 2 j-blocks of 80 cols, 128 batches
        wmma_gemm_bf16_kernel<256><<<dim3(8, 2, NB), 256, 0, stream>>>(
            W1b, P, U1, TPAD, TPAD,
            (size_t)TPAD * TPAD, (size_t)HID * TPAD);

        spike_scan_kernel<<<(NB * HID + 255) / 256, 256, 0, stream>>>(
            U1, S1, NB * HID);

        // V[n] = W2b(32x1024) @ S1[n](1024x160): 2 waves/block
        wmma_gemm_bf16_kernel<64><<<dim3(1, 2, NB), 64, 0, stream>>>(
            W2b, S1, V, HID, TPAD,
            (size_t)HID * TPAD, (size_t)MPAD2 * TPAD);

        psp_spike_out_kernel<<<dim3(NOUT, NB), TPAD, 0, stream>>>(
            V, out, path ? TSAMP : 0);
    }
}
